// Convolution_90426241450688
// MI455X (gfx1250) — compile-verified, matrix operand order checked by reading, not run on hardware
//
#include <hip/hip_runtime.h>

typedef __attribute__((ext_vector_type(16))) _Float16 v16h;
typedef __attribute__((ext_vector_type(8)))  float    v8f;

#define N_      384
#define R_      32
#define H_      32
#define C_      128
#define TILE    16
#define NTILES  (N_ / TILE)          // 24
#define OUTSZ   (N_ * C_ * 11)       // 540672 floats

// output section offsets (concatenated o1..o5, flat, in return order)
#define O1_OFF  0
#define O2_OFF  (N_ * C_)                    // 49152
#define O3_OFF  (O2_OFF + 3 * N_ * C_)       // 196608
#define O4_OFF  (O3_OFF + 3 * N_ * C_)       // 344064
#define O5_OFF  (O4_OFF + N_ * C_)           // 393216

// double-buffered LDS staging area layout (bytes)
#define IMG_OFF 0            // image tile: 16x32 f32  = 2048 B
#define T0_OFF  2048         // t0 tile:    16x128 f32 = 8192 B
#define T1_OFF  10240        // t1 tile:    16x128x3   = 24576 B
#define STAGE_B 34816

// K index held by vector element e for a 16-bit 16x32 A/B fragment (wave32):
// lanes 0-15 (half=0): elems 0..7 -> K 0..7,  elems 8..15 -> K 16..23
// lanes 16-31 (half=1): elems 0..7 -> K 8..15, elems 8..15 -> K 24..31
__device__ __forceinline__ int k_of(int e, int half) {
    return (e < 8) ? (e + 8 * half) : (e + 8 + 8 * half);
}

__global__ __launch_bounds__(128) void conv_fused_kernel(
    const float* __restrict__ image,   // [N,N,R]
    const float* __restrict__ vectors, // [N,N,3]
    const float* __restrict__ t0,      // [N,C,1]
    const float* __restrict__ t1,      // [N,C,3]
    const float* __restrict__ W1,      // [2,2,R,H]
    const float* __restrict__ b1,      // [2,2,H]
    const float* __restrict__ W2,      // [2,2,H,C]
    const float* __restrict__ b2,      // [2,2,C]
    float* __restrict__ partials,      // [S][OUTSZ]
    int tiles_per_split)
{
    const int a     = blockIdx.x;
    const int split = blockIdx.y;
    const int tid   = threadIdx.x;
    const int wave  = tid >> 5;          // 0..3  -> (ti, n)
    const int lane  = tid & 31;
    const int half  = lane >> 4;
    const int l16   = lane & 15;
    const int mlp   = wave;              // ti*2 + n == wave index

    __shared__ alignas(16) unsigned char stage[2][STAGE_B];   // double buffer
    __shared__ _Float16 h_lds[4][TILE * H_];                  // per-wave hidden tile
    __shared__ float    u_lds[2][TILE][4];                    // unit vectors per buffer

    // ---- preload weight fragments (B matrices, K=lane-elem, N=channel) ----
    v16h  w1f[2], w2f[8];
    float b1v[2], b2v[8];
#pragma unroll
    for (int nt = 0; nt < 2; ++nt) {
        const int col = nt * 16 + l16;
#pragma unroll
        for (int e = 0; e < 16; ++e)
            w1f[nt][e] = (_Float16)W1[(mlp * R_ + k_of(e, half)) * H_ + col];
        b1v[nt] = b1[mlp * H_ + col];
    }
#pragma unroll
    for (int nt = 0; nt < 8; ++nt) {
        const int col = nt * 16 + l16;
#pragma unroll
        for (int e = 0; e < 16; ++e)
            w2f[nt][e] = (_Float16)W2[(mlp * H_ + k_of(e, half)) * C_ + col];
        b2v[nt] = b2[mlp * C_ + col];
    }

    float acc[4][8];
#pragma unroll
    for (int i = 0; i < 4; ++i)
#pragma unroll
        for (int j = 0; j < 8; ++j) acc[i][j] = 0.0f;

    // Async stage of one 16-row b-tile into buffer p (ASYNCcnt-tracked, no VGPR
    // round-trip). IOFFSET is applied to BOTH the LDS and global address, so the
    // same voffset register serves every 2KB chunk. Unit vectors go through the
    // normal path (16 lanes, tiny).
    auto issue_stage = [&](int p, int b0) {
        const unsigned lds0 = (unsigned)(size_t)(&stage[p][0]);
        const unsigned vo   = (unsigned)tid * 16u;             // 16 B per lane
        const char* gimg = (const char*)(image + (size_t)(a * N_ + b0) * R_);
        const char* gt0  = (const char*)(t0 + (size_t)b0 * C_);
        const char* gt1  = (const char*)(t1 + (size_t)b0 * C_ * 3);
        asm volatile("global_load_async_to_lds_b128 %0, %1, %2 offset:0"
                     :: "v"(lds0 + IMG_OFF + vo), "v"(vo), "s"(gimg) : "memory");
#pragma unroll
        for (int k = 0; k < 4; ++k)
            asm volatile("global_load_async_to_lds_b128 %0, %1, %2 offset:%3"
                         :: "v"(lds0 + T0_OFF + vo), "v"(vo), "s"(gt0),
                            "i"(k * 2048) : "memory");
#pragma unroll
        for (int k = 0; k < 12; ++k)
            asm volatile("global_load_async_to_lds_b128 %0, %1, %2 offset:%3"
                         :: "v"(lds0 + T1_OFF + vo), "v"(vo), "s"(gt1),
                            "i"(k * 2048) : "memory");
        if (tid < TILE) {
            const int   b  = b0 + tid;
            const float vx = vectors[(size_t)(a * N_ + b) * 3 + 0];
            const float vy = vectors[(size_t)(a * N_ + b) * 3 + 1];
            const float vz = vectors[(size_t)(a * N_ + b) * 3 + 2];
            const float n2 = vx * vx + vy * vy + vz * vz;
            const float inv = (n2 > 0.0f) ? __builtin_amdgcn_rsqf(n2) : 0.0f;
            u_lds[p][tid][0] = vx * inv;
            u_lds[p][tid][1] = vy * inv;
            u_lds[p][tid][2] = vz * inv;
        }
    };

    const int t_beg = split * tiles_per_split;
    const int t_end = t_beg + tiles_per_split;

    issue_stage(0, t_beg * TILE);

    for (int t = t_beg; t < t_end; ++t) {
        const int p = (t - t_beg) & 1;

        asm volatile("s_wait_asynccnt 0x0" ::: "memory");  // this wave's copies done
        __syncthreads();                                   // all waves' copies visible

        if (t + 1 < t_end) issue_stage(p ^ 1, (t + 1) * TILE);  // prefetch next tile

        const float* img_f = (const float*)(stage[p] + IMG_OFF);
        const float* t0_f  = (const float*)(stage[p] + T0_OFF);
        const float* t1_f  = (const float*)(stage[p] + T1_OFF);

        // ---- A fragment from image tile (shared by all 4 waves) ----
        v16h aimg;
#pragma unroll
        for (int e = 0; e < 16; ++e)
            aimg[e] = (_Float16)img_f[l16 * R_ + k_of(e, half)];

        // ---- layer 1: [16x32] x [32x32] + bias + relu -> LDS (f16) ----
#pragma unroll
        for (int nt = 0; nt < 2; ++nt) {
            v8f c0 = {};
            v8f hacc = __builtin_amdgcn_wmma_f32_16x16x32_f16(
                false, aimg, false, w1f[nt], (short)0, c0, false, false);
            const int col = nt * 16 + l16;
#pragma unroll
            for (int v = 0; v < 8; ++v) {
                float hv = hacc[v] + b1v[nt];
                hv = hv > 0.0f ? hv : 0.0f;
                h_lds[wave][(v + 8 * half) * H_ + col] = (_Float16)hv;
            }
        }
        __syncthreads();   // h produced (D layout) -> consumed (A layout)

        // ---- A2 fragment from this wave's hidden tile ----
        v16h a2;
#pragma unroll
        for (int e = 0; e < 16; ++e)
            a2[e] = h_lds[wave][l16 * H_ + k_of(e, half)];

        // ---- layer 2 + fused CG contraction over b ----
        // D layout: lane holds channel c = nt*16+l16; elem v holds b-row m = v+8*half
#pragma unroll
        for (int nt = 0; nt < 8; ++nt) {
            v8f c0 = {};
            v8f d = __builtin_amdgcn_wmma_f32_16x16x32_f16(
                false, a2, false, w2f[nt], (short)0, c0, false, false);
            const int c = nt * 16 + l16;
#pragma unroll
            for (int v = 0; v < 8; ++v) {
                const int   m  = v + 8 * half;
                const float Rv = d[v] + b2v[nt];
                if (wave == 0) {                       // o1 = sum_b R00 * t0
                    acc[0][nt] += Rv * t0_f[m * C_ + c];
                } else if (wave == 1) {                // o2_i = sum_b R01*t0*u_i
                    const float s = Rv * t0_f[m * C_ + c];
                    acc[0][nt] += s * u_lds[p][m][0];
                    acc[1][nt] += s * u_lds[p][m][1];
                    acc[2][nt] += s * u_lds[p][m][2];
                } else if (wave == 2) {                // o3_i = sum_b R10*t1_i
                    const float* tp = &t1_f[(m * C_ + c) * 3];
                    acc[0][nt] += Rv * tp[0];
                    acc[1][nt] += Rv * tp[1];
                    acc[2][nt] += Rv * tp[2];
                } else {                               // o4 = R11*(u.t1), o5 = R11*(u x t1)
                    const float* tp = &t1_f[(m * C_ + c) * 3];
                    const float u0 = u_lds[p][m][0], u1 = u_lds[p][m][1], u2 = u_lds[p][m][2];
                    const float x = tp[0], y = tp[1], z = tp[2];
                    acc[3][nt] += Rv * (u0 * x + u1 * y + u2 * z);
                    acc[0][nt] += Rv * (u1 * z - u2 * y);
                    acc[1][nt] += Rv * (u2 * x - u0 * z);
                    acc[2][nt] += Rv * (u0 * y - u1 * x);
                }
            }
        }
        // no trailing barrier: next iteration's async writes target the other
        // buffer and are issued only after the next top-of-loop barrier
    }

    // ---- combine lane halves (b-rows 0..7 with 8..15), store partials ----
#pragma unroll
    for (int i = 0; i < 4; ++i)
#pragma unroll
        for (int j = 0; j < 8; ++j)
            acc[i][j] += __shfl_xor(acc[i][j], 16, 32);

    float* outp = partials + (size_t)split * OUTSZ;
    if (half == 0) {
#pragma unroll
        for (int nt = 0; nt < 8; ++nt) {
            const int c   = nt * 16 + l16;
            const int afc = a * C_ + c;
            if (wave == 0) {
                outp[O1_OFF + afc] = acc[0][nt];
            } else if (wave == 1) {
                outp[O2_OFF + afc * 3 + 0] = acc[0][nt];
                outp[O2_OFF + afc * 3 + 1] = acc[1][nt];
                outp[O2_OFF + afc * 3 + 2] = acc[2][nt];
            } else if (wave == 2) {
                outp[O3_OFF + afc * 3 + 0] = acc[0][nt];
                outp[O3_OFF + afc * 3 + 1] = acc[1][nt];
                outp[O3_OFF + afc * 3 + 2] = acc[2][nt];
            } else {
                outp[O4_OFF + afc]         = acc[3][nt];
                outp[O5_OFF + afc * 3 + 0] = acc[0][nt];
                outp[O5_OFF + afc * 3 + 1] = acc[1][nt];
                outp[O5_OFF + afc * 3 + 2] = acc[2][nt];
            }
        }
    }
}

__global__ __launch_bounds__(256) void reduce_partials_kernel(
    const float* __restrict__ ws, float* __restrict__ out, int S)
{
    const int i = blockIdx.x * blockDim.x + threadIdx.x;
    if (i < OUTSZ) {
        float s = 0.0f;
        for (int k = 0; k < S; ++k) s += ws[(size_t)k * OUTSZ + i];
        out[i] = s;
    }
}

extern "C" void kernel_launch(void* const* d_in, const int* in_sizes, int n_in,
                              void* d_out, int out_size, void* d_ws, size_t ws_size,
                              hipStream_t stream) {
    const float* image   = (const float*)d_in[0];
    const float* vectors = (const float*)d_in[1];
    const float* t0      = (const float*)d_in[2];
    const float* t1      = (const float*)d_in[3];
    const float* W1      = (const float*)d_in[4];
    const float* b1      = (const float*)d_in[5];
    const float* W2      = (const float*)d_in[6];
    const float* b2      = (const float*)d_in[7];

    // Split the b-reduction across blockIdx.y for parallelism; partial sums go
    // to workspace and are reduced deterministically. Fall back to S=1 writing
    // d_out directly if the workspace is too small.
    int S = 1;
    const int cands[6] = {12, 8, 6, 4, 3, 2};   // all divide NTILES=24
    for (int i = 0; i < 6; ++i) {
        if ((size_t)cands[i] * OUTSZ * sizeof(float) <= ws_size) { S = cands[i]; break; }
    }
    float* partials = (S > 1) ? (float*)d_ws : (float*)d_out;

    dim3 grid(N_, S);
    conv_fused_kernel<<<grid, 128, 0, stream>>>(image, vectors, t0, t1,
                                                W1, b1, W2, b2,
                                                partials, NTILES / S);
    if (S > 1) {
        reduce_partials_kernel<<<(OUTSZ + 255) / 256, 256, 0, stream>>>(
            (const float*)d_ws, (float*)d_out, S);
    }
}
